// RelativeMultiheadAttention_11398843203964
// MI455X (gfx1250) — matrix operand order repaired and checked by reading, not compiled, not run on hardware
//
#include <hip/hip_runtime.h>
#include <stdint.h>

// ---------------------------------------------------------------------------
// Relative multi-head attention forward for MI455X (gfx1250, wave32, WMMA).
// GEMMs: v_wmma_f32_16x16x32_bf16 (bf16 operands, fp32 accumulate).
// Projection GEMMs use a double-buffered GLOBAL_LOAD_ASYNC_TO_LDS_B128
// pipeline (ASYNCcnt) feeding WMMA fragments from LDS.
// ---------------------------------------------------------------------------

#define Bb   4
#define Hh   12
#define Ss   1024
#define Dd   768
#define DHd  64
#define RP   144   // rel_all N-dim padded (129 -> 9 tiles of 16)
#define RW   160   // w K-dim padded      (129 -> 5 steps of 32)
#define BS   (Bb * Ss)          // 4096 rows for projections
#define BH   (Bb * Hh)          // 48 head-batches

typedef __attribute__((ext_vector_type(16))) __bf16 v16bf;
typedef __attribute__((ext_vector_type(8)))  float  v8f;

#define DEV __device__ __forceinline__

DEV uint16_t f2bf(float f) {
  union { float f; uint32_t u; } c; c.f = f;
  uint32_t u = c.u;
  // round-to-nearest-even truncation to bf16
  uint32_t r = (u + 0x7FFFu + ((u >> 16) & 1u)) >> 16;
  return (uint16_t)r;
}

union FragB16 { v16bf v; uint32_t u[8]; };

// A fragment: 16x32 bf16, row-major source, leading dim `ld` (elements, even).
// ISA layout: lanes 0-15 = M rows with K 0-7 / 16-23; lanes 16-31 same M with
// K 8-15 / 24-31; each VGPR packs 2 consecutive K.
DEV v16bf load_frag_a(const uint16_t* base, int ld) {
  const int lane = threadIdx.x & 31;
  const int m = lane & 15, half = lane >> 4;
  FragB16 f;
  const uint32_t* p0 = (const uint32_t*)(base + m * ld + half * 8);
  const uint32_t* p1 = (const uint32_t*)(base + m * ld + 16 + half * 8);
#pragma unroll
  for (int v = 0; v < 4; ++v) { f.u[v] = p0[v]; f.u[4 + v] = p1[v]; }
  return f.v;
}

// Same A fragment but sourced from fp32 data, converted to bf16 on the fly
// (used so attn never needs a bf16 copy in memory).
DEV v16bf load_frag_a_f32(const float* base, int ld) {
  const int lane = threadIdx.x & 31;
  const int m = lane & 15, half = lane >> 4;
  FragB16 f;
  const float* p0 = base + m * ld + half * 8;
  const float* p1 = base + m * ld + 16 + half * 8;
#pragma unroll
  for (int v = 0; v < 4; ++v) {
    uint32_t lo = f2bf(p0[2 * v]), hi = f2bf(p0[2 * v + 1]);
    f.u[v] = lo | (hi << 16);
    lo = f2bf(p1[2 * v]); hi = f2bf(p1[2 * v + 1]);
    f.u[4 + v] = lo | (hi << 16);
  }
  return f.v;
}

// B fragment: 32x16 bf16 where column n of B is row n of a row-major [N][K]
// source (leading dim `ld`). ISA: lane = N, lanes 0-15 K 0-7 / 16-23, lanes 16-31 K 8-15 / 24-31.
DEV v16bf load_frag_b(const uint16_t* base, int ld) {
  const int lane = threadIdx.x & 31;
  const int n = lane & 15, half = lane >> 4;
  FragB16 f;
  const uint32_t* p0 = (const uint32_t*)(base + n * ld + half * 8);
  const uint32_t* p1 = (const uint32_t*)(base + n * ld + 16 + half * 8);
#pragma unroll
  for (int v = 0; v < 4; ++v) { f.u[v] = p0[v]; f.u[4 + v] = p1[v]; }
  return f.v;
}

DEV v8f wmma_bf16(v16bf a, v16bf b, v8f c) {
  return __builtin_amdgcn_wmma_f32_16x16x32_bf16(
      /*neg_a=*/false, a, /*neg_b=*/false, b,
      /*c_mod=*/(short)0, c, /*reuse_a=*/false, /*reuse_b=*/false);
}

// -------- CDNA5 async copy: global -> LDS, 16B per lane, ASYNCcnt tracked ---
// lds_addr: raw LDS byte address (low 32 bits of a generic __shared__ pointer,
// per the ISA flat-address aperture layout). GVS mode: saddr base + v offset.
DEV void async_b128(uint32_t lds_addr, const void* gbase, uint32_t goff) {
  asm volatile("global_load_async_to_lds_b128 %0, %1, %2"
               :: "v"(lds_addr), "v"(goff), "s"(gbase)
               : "memory");
}

DEV void wait_async_le(int n) {
  if (n == 0)
    asm volatile("s_wait_asynccnt 0x0" ::: "memory");
  else
    asm volatile("s_wait_asynccnt 0x3" ::: "memory");
}

// ---------------------------------------------------------------------------
// Elementwise fp32 -> bf16 conversion
// ---------------------------------------------------------------------------
__global__ __launch_bounds__(256) void cvt_bf16(const float* __restrict__ in,
                                                uint16_t* __restrict__ out, int n) {
  for (int i = blockIdx.x * 256 + threadIdx.x; i < n; i += gridDim.x * 256)
    out[i] = f2bf(in[i]);
}

// embK [129,768] -> embKt [12][144][64] bf16 (zero padded), K=DH contiguous
__global__ __launch_bounds__(256) void prep_embK(const float* __restrict__ embK,
                                                 uint16_t* __restrict__ out) {
  int i = blockIdx.x * 256 + threadIdx.x;
  if (i >= Hh * RP * DHd) return;
  int h = i / (RP * DHd);
  int r = (i / DHd) % RP;
  int d = i % DHd;
  float v = (r < 129) ? embK[r * Dd + h * DHd + d] : 0.0f;
  out[i] = f2bf(v);
}

// embV [129,768] -> embVt [12][64][160] bf16 (zero padded), K=r contiguous
__global__ __launch_bounds__(256) void prep_embV(const float* __restrict__ embV,
                                                 uint16_t* __restrict__ out) {
  int i = blockIdx.x * 256 + threadIdx.x;
  if (i >= Hh * DHd * RW) return;
  int h = i / (DHd * RW);
  int d = (i / RW) % DHd;
  int r = i % RW;
  float v = (r < 129) ? embV[r * Dd + h * DHd + d] : 0.0f;
  out[i] = f2bf(v);
}

// ---------------------------------------------------------------------------
// Projection GEMM: C[4096,768] = X[4096,768] * W[768,768]^T + bias
// Block tile 128x64, 8 waves (4x2), each wave 32x32 via 2x2 WMMA, K step 32.
// Operand tiles staged in LDS by a double-buffered async-copy pipeline:
// per stage each wave issues 3 global_load_async_to_lds_b128 (512B each,
// 12KB tile across 8 waves), s_wait_asynccnt 3 overlaps copy with WMMA.
// mode 0: out bf16 head-split [b,h,s,d]             (Q, K)
// mode 1: out bf16 head-split transposed [b,h,d,s]  (V, ready as WMMA-B)
// mode 2: out fp32 row-major [4096,768]             (final output projection)
// ---------------------------------------------------------------------------
__global__ __launch_bounds__(256) void gemm_proj(
    const uint16_t* __restrict__ X, const uint16_t* __restrict__ W,
    const float* __restrict__ bias, uint16_t* __restrict__ out_bf,
    float* __restrict__ out_f, int mode) {
  __shared__ uint16_t tA[2][128][32];   // 2 x 8 KB
  __shared__ uint16_t tB[2][64][32];    // 2 x 4 KB

  const int mt = blockIdx.x, nt = blockIdx.y;
  const int t = threadIdx.x;
  const int wave = t >> 5;
  const int wq = wave & 3, wn = wave >> 2;

  // async-copy chunk assignment (16B chunks; A: 512 chunks, B: 256 chunks)
  const int ra0 = t >> 2,        sa = t & 3;   // A rows 0..63
  const int ra1 = 64 + (t >> 2);               // A rows 64..127
  const int rb  = t >> 2;                      // B rows 0..63
  const uint32_t gA0 = (uint32_t)((mt * 128 + ra0) * Dd + sa * 8) * 2;
  const uint32_t gA1 = (uint32_t)((mt * 128 + ra1) * Dd + sa * 8) * 2;
  const uint32_t gB  = (uint32_t)((nt * 64  + rb ) * Dd + sa * 8) * 2;

  v8f acc[2][2] = {};

  // prologue: stage 0 into buffer 0
  {
    async_b128((uint32_t)(uintptr_t)&tA[0][ra0][sa * 8], X, gA0);
    async_b128((uint32_t)(uintptr_t)&tA[0][ra1][sa * 8], X, gA1);
    async_b128((uint32_t)(uintptr_t)&tB[0][rb ][sa * 8], W, gB);
  }

  const int NK = Dd / 32;   // 24 k-stages
  for (int ks = 0; ks < NK; ++ks) {
    const int buf = ks & 1;
    if (ks + 1 < NK) {
      const uint32_t kb = (uint32_t)(ks + 1) * 64;   // 32 elements * 2B
      async_b128((uint32_t)(uintptr_t)&tA[buf ^ 1][ra0][sa * 8], X, gA0 + kb);
      async_b128((uint32_t)(uintptr_t)&tA[buf ^ 1][ra1][sa * 8], X, gA1 + kb);
      async_b128((uint32_t)(uintptr_t)&tB[buf ^ 1][rb ][sa * 8], W, gB + kb);
      wait_async_le(3);       // stage ks complete; next stage stays in flight
    } else {
      wait_async_le(0);
    }
    __syncthreads();          // tile visible to all waves

    v16bf a0 = load_frag_a(&tA[buf][wq * 32][0], 32);
    v16bf a1 = load_frag_a(&tA[buf][wq * 32 + 16][0], 32);
    v16bf b0 = load_frag_b(&tB[buf][wn * 32][0], 32);
    v16bf b1 = load_frag_b(&tB[buf][wn * 32 + 16][0], 32);
    acc[0][0] = wmma_bf16(a0, b0, acc[0][0]);
    acc[0][1] = wmma_bf16(a0, b1, acc[0][1]);
    acc[1][0] = wmma_bf16(a1, b0, acc[1][0]);
    acc[1][1] = wmma_bf16(a1, b1, acc[1][1]);

    __syncthreads();          // all reads done before buffer is re-filled
  }

  const int m0 = mt * 128 + wq * 32;
  const int n0 = nt * 64 + wn * 32;
  const int lane = threadIdx.x & 31, nl = lane & 15, half = lane >> 4;
#pragma unroll
  for (int mi = 0; mi < 2; ++mi)
#pragma unroll
    for (int ni = 0; ni < 2; ++ni)
#pragma unroll
      for (int r = 0; r < 8; ++r) {
        int row = m0 + mi * 16 + half * 8 + r;   // 0..4095
        int col = n0 + ni * 16 + nl;             // 0..767
        float val = acc[mi][ni][r] + bias[col];
        if (mode == 2) {
          out_f[(size_t)row * Dd + col] = val;
        } else {
          int b = row >> 10, s = row & 1023, h = col >> 6, d = col & 63;
          uint16_t bv = f2bf(val);
          if (mode == 0)
            out_bf[(((size_t)b * Hh + h) * Ss + s) * DHd + d] = bv;
          else
            out_bf[(((size_t)b * Hh + h) * DHd + d) * Ss + s] = bv;
        }
      }
}

// ---------------------------------------------------------------------------
// rel_all[bh,q,r] = Q[bh,q,:] . embK_h[r,:]   (N padded to 144, fp32 out)
// ---------------------------------------------------------------------------
__global__ __launch_bounds__(256) void rel_scores(
    const uint16_t* __restrict__ Qbf, const uint16_t* __restrict__ embKt,
    float* __restrict__ rel) {
  const int qt = blockIdx.x;          // 8
  const int bh = blockIdx.y;          // 48
  const int h = bh % Hh;
  const int wave = threadIdx.x >> 5;
  const int q0 = qt * 128 + wave * 16;
  const uint16_t* Abase = Qbf + ((size_t)bh * Ss + q0) * DHd;
  const int lane = threadIdx.x & 31, nl = lane & 15, half = lane >> 4;
#pragma unroll
  for (int nt = 0; nt < RP / 16; ++nt) {
    v8f acc = {};
    const uint16_t* Bbase = embKt + ((size_t)h * RP + nt * 16) * DHd;
#pragma unroll
    for (int k = 0; k < DHd; k += 32)
      acc = wmma_bf16(load_frag_a(Abase + k, DHd),
                      load_frag_b(Bbase + k, DHd), acc);
#pragma unroll
    for (int r = 0; r < 8; ++r) {
      int q = q0 + half * 8 + r;
      rel[((size_t)bh * Ss + q) * RP + nt * 16 + nl] = acc[r];
    }
  }
}

// ---------------------------------------------------------------------------
// scores = (Q K^T + rel gather) / sqrt(DH), written fp32 into attn region.
// Block tile 128(q) x 64(k) per (b,h); 8 waves (4x2) of 32x32.
// Q/K head tiles are 128KB -> L2-resident, direct global fragments.
// ---------------------------------------------------------------------------
__global__ __launch_bounds__(256) void qk_scores(
    const uint16_t* __restrict__ Qbf, const uint16_t* __restrict__ Kbf,
    const float* __restrict__ rel, float* __restrict__ attn) {
  const int qt = blockIdx.x, kt = blockIdx.y, bh = blockIdx.z;
  const int wave = threadIdx.x >> 5;
  const int wq = wave & 3, wk = wave >> 2;
  const int q0 = qt * 128 + wq * 32;
  const int k0 = kt * 64 + wk * 32;
  const uint16_t* Qb = Qbf + ((size_t)bh * Ss + q0) * DHd;
  const uint16_t* Kb = Kbf + ((size_t)bh * Ss + k0) * DHd;

  v8f acc[2][2] = {};
#pragma unroll
  for (int d = 0; d < DHd; d += 32) {
    v16bf a0 = load_frag_a(Qb + d, DHd);
    v16bf a1 = load_frag_a(Qb + 16 * DHd + d, DHd);
    v16bf b0 = load_frag_b(Kb + d, DHd);
    v16bf b1 = load_frag_b(Kb + 16 * DHd + d, DHd);
    acc[0][0] = wmma_bf16(a0, b0, acc[0][0]);
    acc[0][1] = wmma_bf16(a0, b1, acc[0][1]);
    acc[1][0] = wmma_bf16(a1, b0, acc[1][0]);
    acc[1][1] = wmma_bf16(a1, b1, acc[1][1]);
  }

  const int lane = threadIdx.x & 31, nl = lane & 15, half = lane >> 4;
  const float* relrow = rel + (size_t)bh * Ss * RP;
  float* arow = attn + (size_t)bh * Ss * Ss;
#pragma unroll
  for (int mi = 0; mi < 2; ++mi)
#pragma unroll
    for (int ni = 0; ni < 2; ++ni)
#pragma unroll
      for (int r = 0; r < 8; ++r) {
        int q = q0 + mi * 16 + half * 8 + r;
        int k = k0 + ni * 16 + nl;
        int rr = k - q;
        rr = rr < -64 ? -64 : (rr > 64 ? 64 : rr);
        float sc = (acc[mi][ni][r] + relrow[(size_t)q * RP + rr + 64]) * 0.125f;
        arow[(size_t)q * Ss + k] = sc;
      }
}

// ---------------------------------------------------------------------------
// Row softmax over k (in place, fp32 in d_out) + band scatter into w[q,r]
// (bf16, K padded to 160). Interior bins have a unique writer; the two edge
// bins (r=0, r=128) accumulate with LDS ds_add_f32 atomics.
// ---------------------------------------------------------------------------
__global__ __launch_bounds__(256) void softmax_w(float* __restrict__ attn,
                                                 uint16_t* __restrict__ wbf) {
  const int row = blockIdx.x;           // bh*1024 + q
  const int q = row & (Ss - 1);
  float* a = attn + (size_t)row * Ss;
  const int t = threadIdx.x;

  __shared__ float red[8];
  __shared__ float wrow[RW];

  float v[4];
  float mx = -3.4e38f;
#pragma unroll
  for (int i = 0; i < 4; ++i) { v[i] = a[t + i * 256]; mx = fmaxf(mx, v[i]); }
#pragma unroll
  for (int off = 16; off; off >>= 1) mx = fmaxf(mx, __shfl_xor(mx, off, 32));
  if ((t & 31) == 0) red[t >> 5] = mx;
  __syncthreads();
  mx = red[0];
#pragma unroll
  for (int i = 1; i < 8; ++i) mx = fmaxf(mx, red[i]);
  __syncthreads();

  float sum = 0.0f;
#pragma unroll
  for (int i = 0; i < 4; ++i) { v[i] = __expf(v[i] - mx); sum += v[i]; }
#pragma unroll
  for (int off = 16; off; off >>= 1) sum += __shfl_xor(sum, off, 32);
  if ((t & 31) == 0) red[t >> 5] = sum;
  if (t < RW) wrow[t] = 0.0f;
  __syncthreads();
  sum = 0.0f;
#pragma unroll
  for (int i = 0; i < 8; ++i) sum += red[i];
  const float inv = 1.0f / sum;

#pragma unroll
  for (int i = 0; i < 4; ++i) {
    int k = t + i * 256;
    float p = v[i] * inv;
    a[k] = p;
    int rr = k - q;
    if (rr < -64)      atomicAdd(&wrow[0], p);
    else if (rr > 64)  atomicAdd(&wrow[128], p);
    else               wrow[rr + 64] = p;       // unique writer per row
  }
  __syncthreads();
  if (t < RW) wbf[(size_t)row * RW + t] = f2bf(wrow[t]);
}

// ---------------------------------------------------------------------------
// ctx = attn @ V + w @ embV_h, stored bf16 in [b, s, h*64+d] layout.
// attn is fp32 (from d_out) converted to bf16 inside the fragment loader
// (201MB stream is touched exactly once here). V^T / embV^T are WMMA-B ready.
// Per block: 128 q rows x 64 d cols for one (b,h); 8 waves of 16x64.
// ---------------------------------------------------------------------------
__global__ __launch_bounds__(256) void attn_v(
    const float* __restrict__ attn, const uint16_t* __restrict__ Vt,
    const uint16_t* __restrict__ wbf, const uint16_t* __restrict__ embVt,
    uint16_t* __restrict__ ctx) {
  const int qt = blockIdx.x, bh = blockIdx.y;
  const int h = bh % Hh, b = bh / Hh;
  const int wave = threadIdx.x >> 5;
  const int q0 = qt * 128 + wave * 16;

  const float* Ab = attn + ((size_t)bh * Ss + q0) * Ss;
  const uint16_t* Vb = Vt + (size_t)bh * DHd * Ss;   // [d][s], ld = 1024

  v8f acc[4] = {};
  for (int k = 0; k < Ss; k += 32) {
    v16bf a = load_frag_a_f32(Ab + k, Ss);
#pragma unroll
    for (int nn = 0; nn < 4; ++nn)
      acc[nn] = wmma_bf16(a, load_frag_b(Vb + (size_t)(nn * 16) * Ss + k, Ss),
                          acc[nn]);
  }

  const uint16_t* Wb = wbf + ((size_t)bh * Ss + q0) * RW;
  const uint16_t* Eb = embVt + (size_t)h * DHd * RW;  // [d][r], ld = 160
#pragma unroll
  for (int k = 0; k < RW; k += 32) {
    v16bf a = load_frag_a(Wb + k, RW);
#pragma unroll
    for (int nn = 0; nn < 4; ++nn)
      acc[nn] = wmma_bf16(a, load_frag_b(Eb + (size_t)(nn * 16) * RW + k, RW),
                          acc[nn]);
  }

  const int lane = threadIdx.x & 31, nl = lane & 15, half = lane >> 4;
#pragma unroll
  for (int nn = 0; nn < 4; ++nn)
#pragma unroll
    for (int r = 0; r < 8; ++r) {
      int qq = q0 + half * 8 + r;
      int c = h * DHd + nn * 16 + nl;
      ctx[((size_t)b * Ss + qq) * Dd + c] = f2bf(acc[nn][r]);
    }
}

// ---------------------------------------------------------------------------
// Host-side launcher
// ---------------------------------------------------------------------------
extern "C" void kernel_launch(void* const* d_in, const int* in_sizes, int n_in,
                              void* d_out, int out_size, void* d_ws, size_t ws_size,
                              hipStream_t stream) {
  (void)in_sizes; (void)n_in; (void)out_size; (void)ws_size;

  const float* q_in  = (const float*)d_in[0];
  const float* k_in  = (const float*)d_in[1];
  const float* v_in  = (const float*)d_in[2];
  const float* WQw   = (const float*)d_in[3];
  const float* WQb   = (const float*)d_in[4];
  const float* WKw   = (const float*)d_in[5];
  const float* WKb   = (const float*)d_in[6];
  const float* WVw   = (const float*)d_in[7];
  const float* WVb   = (const float*)d_in[8];
  const float* WOw   = (const float*)d_in[9];
  const float* WOb   = (const float*)d_in[10];
  const float* embK  = (const float*)d_in[11];
  const float* embV  = (const float*)d_in[12];

  float* out_f = (float*)d_out;                         // [4,1024,768]
  float* attn  = out_f + (size_t)Bb * Ss * Dd;          // [4,12,1024,1024]

  // workspace carve-out (256B aligned slices)
  size_t off = 0;
  auto take = [&](size_t bytes) {
    size_t o = off;
    off = (off + bytes + 255) & ~(size_t)255;
    return (char*)d_ws + o;
  };
  uint16_t* Xq    = (uint16_t*)take((size_t)BS * Dd * 2);
  uint16_t* Xk    = (uint16_t*)take((size_t)BS * Dd * 2);
  uint16_t* Xv    = (uint16_t*)take((size_t)BS * Dd * 2);
  uint16_t* Wq    = (uint16_t*)take((size_t)Dd * Dd * 2);
  uint16_t* Wk    = (uint16_t*)take((size_t)Dd * Dd * 2);
  uint16_t* Wv    = (uint16_t*)take((size_t)Dd * Dd * 2);
  uint16_t* Wo    = (uint16_t*)take((size_t)Dd * Dd * 2);
  uint16_t* embKt = (uint16_t*)take((size_t)Hh * RP * DHd * 2);
  uint16_t* embVt = (uint16_t*)take((size_t)Hh * DHd * RW * 2);
  uint16_t* Qbf   = (uint16_t*)take((size_t)BH * Ss * DHd * 2);
  uint16_t* Kbf   = (uint16_t*)take((size_t)BH * Ss * DHd * 2);
  uint16_t* Vt    = (uint16_t*)take((size_t)BH * DHd * Ss * 2);
  float*    rel   = (float*)   take((size_t)BH * Ss * RP * 4);
  uint16_t* wbf   = (uint16_t*)take((size_t)BH * Ss * RW * 2);
  uint16_t* ctx   = (uint16_t*)take((size_t)BS * Dd * 2);

  const int nAct = BS * Dd;        // 3,145,728
  const int nW   = Dd * Dd;        // 589,824

  // 1) fp32 -> bf16 conversions
  cvt_bf16<<<2048, 256, 0, stream>>>(q_in, Xq, nAct);
  cvt_bf16<<<2048, 256, 0, stream>>>(k_in, Xk, nAct);
  cvt_bf16<<<2048, 256, 0, stream>>>(v_in, Xv, nAct);
  cvt_bf16<<<1024, 256, 0, stream>>>(WQw, Wq, nW);
  cvt_bf16<<<1024, 256, 0, stream>>>(WKw, Wk, nW);
  cvt_bf16<<<1024, 256, 0, stream>>>(WVw, Wv, nW);
  cvt_bf16<<<1024, 256, 0, stream>>>(WOw, Wo, nW);
  prep_embK<<<(Hh * RP * DHd + 255) / 256, 256, 0, stream>>>(embK, embKt);
  prep_embV<<<(Hh * DHd * RW + 255) / 256, 256, 0, stream>>>(embV, embVt);

  // 2) input projections (WMMA + async LDS pipeline)
  dim3 gp(BS / 128, Dd / 64);
  gemm_proj<<<gp, 256, 0, stream>>>(Xq, Wq, WQb, Qbf, nullptr, 0);
  gemm_proj<<<gp, 256, 0, stream>>>(Xk, Wk, WKb, Kbf, nullptr, 0);
  gemm_proj<<<gp, 256, 0, stream>>>(Xv, Wv, WVb, Vt,  nullptr, 1);

  // 3) relative position scores (WMMA)
  rel_scores<<<dim3(Ss / 128, BH), 256, 0, stream>>>(Qbf, embKt, rel);

  // 4) QK^T + rel gather + scale -> attn region (WMMA)
  qk_scores<<<dim3(Ss / 128, Ss / 64, BH), 256, 0, stream>>>(Qbf, Kbf, rel, attn);

  // 5) softmax (in place) + band scatter to w (LDS ds_add_f32)
  softmax_w<<<BH * Ss, 256, 0, stream>>>(attn, wbf);

  // 6) attn@V + w@embV -> context (WMMA, attn converted bf16 in-loader)
  attn_v<<<dim3(Ss / 128, BH), 256, 0, stream>>>(attn, Vt, wbf, embVt, ctx);

  // 7) output projection -> d_out (WMMA)
  gemm_proj<<<gp, 256, 0, stream>>>(ctx, Wo, WOb, nullptr, out_f, 2);
}
